// GroupCExpertPool_78288663872351
// MI455X (gfx1250) — matrix operand order repaired and checked by reading, not compiled, not run on hardware
//
#include <hip/hip_runtime.h>
#include <hip/hip_bf16.h>
#include <cstdint>
#include <cstddef>

// Problem constants (match reference)
#define TT 2048   // B*N tokens
#define DD 1024   // model dim
#define EE 8      // experts
#define HH 4096   // hidden dim (D*FF_MULT)

// Tiling
#define BM 32     // token rows per block
#define BN 128    // output cols per block
#define KT 32     // K slab per LDS stage (one WMMA K)
#define LSTR 40   // padded LDS row stride (bf16 elements) = 80B

typedef __attribute__((ext_vector_type(16))) __bf16 v16bf;
typedef __attribute__((ext_vector_type(8)))  float  v8f;
typedef __attribute__((ext_vector_type(2)))  float  v2f;
typedef __attribute__((ext_vector_type(2)))  __bf16 v2bf;
typedef __attribute__((ext_vector_type(2)))  int    v2i;

// ---- feature probes -------------------------------------------------------
#if defined(__has_builtin)
#  if __has_builtin(__builtin_amdgcn_global_load_async_to_lds_b64)
#    define HAVE_ASYNC_LDS 1
#  endif
#endif
#ifndef HAVE_ASYNC_LDS
#  define HAVE_ASYNC_LDS 0
#endif

#if HAVE_ASYNC_LDS
// builtin signature: (v2i AS1* src, v2i AS3* dst, imm offset, imm cpol)
typedef __attribute__((address_space(1))) v2i g_v2i;
typedef __attribute__((address_space(3))) v2i l_v2i;
#endif

__device__ __forceinline__ void wait_async0() {
#if defined(__has_builtin) && __has_builtin(__builtin_amdgcn_s_wait_asynccnt)
  __builtin_amdgcn_s_wait_asynccnt(0);
#else
  asm volatile("s_wait_asynccnt 0x0" ::: "memory");
#endif
}

// ---- packed fp32 -> bf16 conversion (v_cvt_pk_bf16_f32) -------------------
__device__ __forceinline__ unsigned pk_bf2(float a, float b) {
  v2f t; t.x = a; t.y = b;
  v2bf h = __builtin_convertvector(t, v2bf);
  return __builtin_bit_cast(unsigned, h);
}

__device__ __forceinline__ unsigned short f2bf(float f) {
  __bf16 h = (__bf16)f;
  return __builtin_bit_cast(unsigned short, h);
}

// pack 4 floats -> 4 bf16 and store as one 8-byte LDS write
__device__ __forceinline__ void st_bf4(unsigned short* p, float a, float b, float c, float d) {
  uint2 v;
  v.x = pk_bf2(a, b);
  v.y = pk_bf2(c, d);
  *(uint2*)p = v;
}

struct Q32 { uint4 lo, hi; };
// assemble a 16-element bf16 WMMA fragment from two 16B LDS chunks
__device__ __forceinline__ v16bf ld_frag(const unsigned short* p0, const unsigned short* p1) {
  Q32 q;
  q.lo = *(const uint4*)p0;
  q.hi = *(const uint4*)p1;
  return __builtin_bit_cast(v16bf, q);
}

__device__ __forceinline__ v8f wmma_bf16(v16bf a, v16bf b, v8f c) {
  // (neg_a, A, neg_b, B, c_mod, C, reuse_a, reuse_b)
  return __builtin_amdgcn_wmma_f32_16x16x32_bf16(false, a, false, b, (short)0, c, false, false);
}

// ---------------------------------------------------------------------------
// Kernel 1: act[t][e][h] = gelu(x@Wg[e]) * (x@Wv[e]),  bf16 output
// grid: (HH/BN, TT/BM, EE), block: 256 (8 waves: 2 M x 4 N)
// ---------------------------------------------------------------------------
__global__ __launch_bounds__(256) void moe_ffn1(const float* __restrict__ x,
                                                const float* __restrict__ Wg,
                                                const float* __restrict__ Wv,
                                                unsigned short* __restrict__ act) {
  const int e  = blockIdx.z;
  const int n0 = blockIdx.x * BN;
  const int t0 = blockIdx.y * BM;
  const float* wg = Wg + (size_t)e * DD * HH;
  const float* wv = Wv + (size_t)e * DD * HH;

  __shared__ __align__(16) unsigned short lA [BM][LSTR];
  __shared__ __align__(16) unsigned short lBg[BN][LSTR];
  __shared__ __align__(16) unsigned short lBv[BN][LSTR];

  const int tid   = threadIdx.x;
  const int lane  = tid & 31;
  const int wave  = tid >> 5;
  const int waveM = wave >> 2;   // 0..1
  const int waveN = wave & 3;    // 0..3
  const int lm    = lane & 15;
  const int kh    = lane >> 4;   // K-half select

  v8f cg[2] = {};
  v8f cv[2] = {};

  const int am = tid >> 3;           // A-stage row   (0..31)
  const int ak = (tid & 7) * 4;      // A-stage k     (0,4,..,28)
  const int bk = (tid >> 5) * 4;     // B-stage k base(0,4,..,28)
  const int bn = (tid & 31) * 4;     // B-stage n base(0,4,..,124)

  for (int kk = 0; kk < DD; kk += KT) {
    // ---- stage A: x tile, fp32 -> bf16, [m][k]
    {
      const float4 v = *(const float4*)(x + (size_t)(t0 + am) * DD + kk + ak);
      st_bf4(&lA[am][ak], v.x, v.y, v.z, v.w);
    }
    // ---- stage B: Wg/Wv tiles, fp32 -> bf16, transposed to [n][k]
    {
      float gb[4][4], vb[4][4];
#pragma unroll
      for (int i = 0; i < 4; ++i) {
        const size_t row = (size_t)(kk + bk + i) * HH + (size_t)(n0 + bn);
        const float4 g = *(const float4*)(wg + row);
        const float4 v = *(const float4*)(wv + row);
        gb[i][0] = g.x; gb[i][1] = g.y; gb[i][2] = g.z; gb[i][3] = g.w;
        vb[i][0] = v.x; vb[i][1] = v.y; vb[i][2] = v.z; vb[i][3] = v.w;
        // speculative prefetch of next K slab (safe past end: silently dropped)
        __builtin_prefetch(wg + row + (size_t)KT * HH, 0, 1);
        __builtin_prefetch(wv + row + (size_t)KT * HH, 0, 1);
      }
#pragma unroll
      for (int j = 0; j < 4; ++j) {
        st_bf4(&lBg[bn + j][bk], gb[0][j], gb[1][j], gb[2][j], gb[3][j]);
        st_bf4(&lBv[bn + j][bk], vb[0][j], vb[1][j], vb[2][j], vb[3][j]);
      }
    }
    __syncthreads();
    // ---- compute: one A frag, 2 N-tiles x 2 matrices
    {
      const int ar = waveM * 16 + lm;
      const v16bf a = ld_frag(&lA[ar][kh * 16], &lA[ar][kh * 16 + 8]);
#pragma unroll
      for (int j = 0; j < 2; ++j) {
        const int br = waveN * 32 + j * 16 + lm;
        const v16bf bg = ld_frag(&lBg[br][kh * 16], &lBg[br][kh * 16 + 8]);
        const v16bf bv = ld_frag(&lBv[br][kh * 16], &lBv[br][kh * 16 + 8]);
        cg[j] = wmma_bf16(a, bg, cg[j]);
        cv[j] = wmma_bf16(a, bv, cv[j]);
      }
    }
    __syncthreads();
  }

  // ---- epilogue: exact GELU(gate) * value -> bf16 act
#pragma unroll
  for (int j = 0; j < 2; ++j) {
#pragma unroll
    for (int r = 0; r < 8; ++r) {
      const float g  = cg[j][r];
      const float gl = 0.5f * g * (1.0f + erff(g * 0.70710678118654752f));
      const float o  = gl * cv[j][r];
      const int m = t0 + waveM * 16 + r + 8 * kh;
      const int n = n0 + waveN * 32 + j * 16 + lm;
      act[((size_t)m * EE + e) * HH + n] = f2bf(o);
    }
  }
}

// ---------------------------------------------------------------------------
// Kernel 2: out[t][d] = sum_e w[t,e] * (act[t,e,:] @ Wo[e])[d]
//           w[t,e] = (disp>0 ? comb : 0) * scale[e]
// grid: (DD/BN, TT/BM), block: 256
// ---------------------------------------------------------------------------
__global__ __launch_bounds__(256) void moe_ffn2(const unsigned short* __restrict__ act,
                                                const float* __restrict__ Wo,
                                                const float* __restrict__ disp,
                                                const float* __restrict__ comb,
                                                const float* __restrict__ scale,
                                                float* __restrict__ out) {
  const int n0 = blockIdx.x * BN;
  const int t0 = blockIdx.y * BM;

  __shared__ __align__(16) unsigned short lA[BM][LSTR];
  __shared__ __align__(16) unsigned short lB[BN][LSTR];
  __shared__ float lw[BM][EE];

  const int tid   = threadIdx.x;
  const int lane  = tid & 31;
  const int wave  = tid >> 5;
  const int waveM = wave >> 2;
  const int waveN = wave & 3;
  const int lm    = lane & 15;
  const int kh    = lane >> 4;

  const int am = tid >> 3;
  const int ak = (tid & 7) * 4;
  const int bk = (tid >> 5) * 4;
  const int bn = (tid & 31) * 4;

  // per-row combine weights (32 rows x 8 experts = 256 threads, one each)
  {
    const int m = tid >> 3, ee = tid & 7;
    const int t = t0 + m;
    const float dv = disp[(size_t)t * EE + ee];
    lw[m][ee] = (dv > 0.0f ? comb[(size_t)t * EE + ee] : 0.0f) * scale[ee];
  }

  float accf[2][8] = {};

  for (int e = 0; e < EE; ++e) {
    const float* wo = Wo + (size_t)e * HH * DD;
    v8f c[2] = {};
    for (int kk = 0; kk < HH; kk += KT) {
      // ---- stage A: act tile (already bf16) -> LDS via async DMA
      {
        const size_t idx = ((size_t)(t0 + am) * EE + e) * HH + kk + ak;
#if HAVE_ASYNC_LDS
        __builtin_amdgcn_global_load_async_to_lds_b64(
            (g_v2i*)(act + idx), (l_v2i*)&lA[am][ak], 0, 0);
#else
        *(uint2*)&lA[am][ak] = *(const uint2*)(act + idx);
#endif
      }
      // ---- stage B: Wo tile, fp32 -> bf16, transposed to [n][k]
      {
        float wb[4][4];
#pragma unroll
        for (int i = 0; i < 4; ++i) {
          const size_t row = (size_t)(kk + bk + i) * DD + (size_t)(n0 + bn);
          const float4 w = *(const float4*)(wo + row);
          wb[i][0] = w.x; wb[i][1] = w.y; wb[i][2] = w.z; wb[i][3] = w.w;
          __builtin_prefetch(wo + row + (size_t)KT * DD, 0, 1);  // speculative
        }
#pragma unroll
        for (int j = 0; j < 4; ++j) {
          st_bf4(&lB[bn + j][bk], wb[0][j], wb[1][j], wb[2][j], wb[3][j]);
        }
      }
#if HAVE_ASYNC_LDS
      wait_async0();
#endif
      __syncthreads();
      {
        const int ar = waveM * 16 + lm;
        const v16bf a = ld_frag(&lA[ar][kh * 16], &lA[ar][kh * 16 + 8]);
#pragma unroll
        for (int j = 0; j < 2; ++j) {
          const int br = waveN * 32 + j * 16 + lm;
          const v16bf b = ld_frag(&lB[br][kh * 16], &lB[br][kh * 16 + 8]);
          c[j] = wmma_bf16(a, b, c[j]);
        }
      }
      __syncthreads();
    }
    // scaled accumulate (row weight depends on C-layout row = r + 8*kh)
#pragma unroll
    for (int j = 0; j < 2; ++j) {
#pragma unroll
      for (int r = 0; r < 8; ++r) {
        accf[j][r] += lw[waveM * 16 + r + 8 * kh][e] * c[j][r];
      }
    }
  }

  // ---- write output (fp32)
#pragma unroll
  for (int j = 0; j < 2; ++j) {
#pragma unroll
    for (int r = 0; r < 8; ++r) {
      const int m = t0 + waveM * 16 + r + 8 * kh;
      const int n = n0 + waveN * 32 + j * 16 + lm;
      out[(size_t)m * DD + n] = accf[j][r];
    }
  }
}

// ---------------------------------------------------------------------------
extern "C" void kernel_launch(void* const* d_in, const int* in_sizes, int n_in,
                              void* d_out, int out_size, void* d_ws, size_t ws_size,
                              hipStream_t stream) {
  (void)in_sizes; (void)n_in; (void)out_size; (void)ws_size;
  const float* tokens = (const float*)d_in[0];   // (B,N,D) fp32
  const float* disp   = (const float*)d_in[1];   // (B,N,E) fp32
  const float* comb   = (const float*)d_in[2];   // (B,N,E) fp32
  const float* Wg     = (const float*)d_in[3];   // (E,D,H) fp32
  const float* Wv     = (const float*)d_in[4];   // (E,D,H) fp32
  const float* Wo     = (const float*)d_in[5];   // (E,H,D) fp32
  const float* scale  = (const float*)d_in[6];   // (E,)   fp32
  float* out = (float*)d_out;

  // workspace: bf16 act[T][E][H] = 2048*8*4096*2 = 128 MB
  unsigned short* act = (unsigned short*)d_ws;

  dim3 blk(256, 1, 1);
  dim3 g1(HH / BN, TT / BM, EE);   // 32 x 64 x 8
  moe_ffn1<<<g1, blk, 0, stream>>>(tokens, Wg, Wv, act);

  dim3 g2(DD / BN, TT / BM, 1);    // 8 x 64
  moe_ffn2<<<g2, blk, 0, stream>>>(act, Wo, disp, comb, scale, out);
}
